// CommNetLSTM_89756226552295
// MI455X (gfx1250) — matrix operand order repaired and checked by reading, not compiled, not run on hardware
//
#include <hip/hip_runtime.h>

// ---------------------------------------------------------------------------
// CommNet 2-hop LSTM for gfx1250 (MI455X).
// Kernel 1 (tiny): convert all weights f32->f16 ONCE into d_ws, pre-swizzled
//   into the WMMA B-fragment layout (32 lanes x 16 f16 per fragment).
// Kernel 2 (main): 1 workgroup = 4 batches = 128 rows, 8 wave32s.
//   Fragments are async-copied (GLOBAL_LOAD_ASYNC_TO_LDS_B128, ASYNCcnt) from
//   d_ws into LDS, GEMMs run as v_wmma_f32_16x16x32_f16 fed by ds_load_b128.
//   x lives in LDS (f32) to keep the kernel under 256 VGPRs.
// ---------------------------------------------------------------------------

typedef __attribute__((ext_vector_type(16))) _Float16 v16h;
typedef __attribute__((ext_vector_type(8)))  _Float16 v8h;
typedef __attribute__((ext_vector_type(4)))  _Float16 v4h;
typedef __attribute__((ext_vector_type(8)))  float    v8f;
typedef __attribute__((ext_vector_type(4)))  float    v4f;
typedef int v4i __attribute__((vector_size(4 * sizeof(int))));

#define B_DIM  4096
#define N_AG   32
#define OBS_D  128
#define H_D    64
#define A_D    16
#define ROWS_PER_WG 128
#define BN     (B_DIM * N_AG)
#define NTHREADS 256

// output layout in d_out (floats): logits | values | h | c
#define OFF_L  0L
#define OFF_V  ((long)BN * A_D)
#define OFF_H  (OFF_V + (long)BN)
#define OFF_C  (OFF_H + (long)BN * H_D)

#define FRAG_ELEMS 512            // one B fragment: 32 lanes x 16 f16

// f16 fragment images in d_ws (element offsets)
#define WS_ENC   0        // 16 frags (128x64)
#define WS_MSG0  8192     //  8 frags (64x64)
#define WS_MSG1  12288    //  8 frags
#define WS_WX0   16384    // 32 frags (64x256)
#define WS_WH0   32768    // 32 frags
#define WS_WX1   49152    // 32 frags
#define WS_WH1   65536    // 32 frags
#define WS_ACT   81920    //  2 frags (64x16)

#if __has_builtin(__builtin_amdgcn_global_load_async_to_lds_b128)
#define ASYNC_LDS_COPY 1
#endif

__device__ __forceinline__ v8f wmma_f16(v16h a, v16h b, v8f c) {
  return __builtin_amdgcn_wmma_f32_16x16x32_f16(
      false, a, false, b, (short)0, c, false, false);
}

__device__ __forceinline__ float sigf(float v) {
  return 1.0f / (1.0f + __expf(-v));
}
__device__ __forceinline__ float tanh_fast(float x) {
  float ax = fabsf(x);
  float e  = __expf(-2.0f * ax);
  float r  = (1.0f - e) / (1.0f + e);
  return copysignf(r, x);
}

// A-fragment (16 rows x 32 K, f16) from an LDS f16 tile (row stride `stride`).
__device__ __forceinline__ v16h load_a_lds(const _Float16* lds, int r0, int k0,
                                           int stride, int lane) {
  v16h a;
  const int m  = r0 + (lane & 15);
  const int kh = (lane >> 4) << 3;
#pragma unroll
  for (int v = 0; v < 8; ++v) {
    int kk = k0 + ((v >> 2) << 4) + kh + ((v & 3) << 1);
    const _Float16* p = lds + m * stride + kk;
    a[2 * v]     = p[0];
    a[2 * v + 1] = p[1];
  }
  return a;
}

// Pre-swizzled B fragment from LDS: lane reads 16 contiguous f16 (2x b128).
__device__ __forceinline__ v16h load_bfrag(const _Float16* base, int f, int lane) {
  return *(const v16h*)(base + f * FRAG_ELEMS + lane * 16);
}

// Copy pre-swizzled fragment bytes from d_ws into LDS (16B per lane per op).
__device__ __forceinline__ void copy_frags(const _Float16* __restrict__ src,
                                           _Float16* dst, int nElems, int tid) {
#ifdef ASYNC_LDS_COPY
  for (int e = tid * 8; e < nElems; e += NTHREADS * 8)
    __builtin_amdgcn_global_load_async_to_lds_b128(
        (__attribute__((address_space(1))) v4i*)(src + e),
        (__attribute__((address_space(3))) v4i*)(dst + e),
        /*offset=*/0, /*cpol=*/0);
#else
  for (int e = tid * 8; e < nElems; e += NTHREADS * 8)
    *(v8h*)(dst + e) = *(const v8h*)(src + e);
#endif
}

__device__ __forceinline__ void async_join() {
#ifdef ASYNC_LDS_COPY
#if __has_builtin(__builtin_amdgcn_s_wait_asynccnt)
  __builtin_amdgcn_s_wait_asynccnt(0);
#else
  asm volatile("s_wait_asynccnt 0x0" ::: "memory");
#endif
#endif
}

// ---------------------------------------------------------------------------
// Kernel 1: one-time weight conversion into pre-swizzled f16 fragments.
// ---------------------------------------------------------------------------
__device__ void stage_w_g(const float* __restrict__ W, int K, int N,
                          _Float16* __restrict__ dst, int tid) {
  const int ntiles = N >> 4;
  for (int idx = tid * 4; idx < K * N; idx += NTHREADS * 4) {
    v4f w = *(const v4f*)(W + idx);
    int k  = idx / N;
    int n0 = idx - k * N;
    int kt = k >> 5, kr = k & 31;
    int lhi = ((kr >> 3) & 1) << 4;
    int j   = (kr & 7) + ((kr >> 4) << 3);
#pragma unroll
    for (int u = 0; u < 4; ++u) {
      int n  = n0 + u;
      int nt = n >> 4, nl = n & 15;
      dst[(kt * ntiles + nt) * FRAG_ELEMS + (lhi + nl) * 16 + j] = (_Float16)w[u];
    }
  }
}

__global__ void __launch_bounds__(256)
stage_weights_kernel(const float* __restrict__ enc_W,
                     const float* __restrict__ msg_W0, const float* __restrict__ msg_W1,
                     const float* __restrict__ Wx0, const float* __restrict__ Wh0,
                     const float* __restrict__ Wx1, const float* __restrict__ Wh1,
                     const float* __restrict__ act_W, _Float16* __restrict__ ws) {
  switch (blockIdx.x) {
    case 0: stage_w_g(enc_W,  OBS_D, H_D,     ws + WS_ENC,  threadIdx.x); break;
    case 1: stage_w_g(msg_W0, H_D,   H_D,     ws + WS_MSG0, threadIdx.x); break;
    case 2: stage_w_g(msg_W1, H_D,   H_D,     ws + WS_MSG1, threadIdx.x); break;
    case 3: stage_w_g(Wx0,    H_D,   4 * H_D, ws + WS_WX0,  threadIdx.x); break;
    case 4: stage_w_g(Wh0,    H_D,   4 * H_D, ws + WS_WH0,  threadIdx.x); break;
    case 5: stage_w_g(Wx1,    H_D,   4 * H_D, ws + WS_WX1,  threadIdx.x); break;
    case 6: stage_w_g(Wh1,    H_D,   4 * H_D, ws + WS_WH1,  threadIdx.x); break;
    default: stage_w_g(act_W, H_D,   A_D,     ws + WS_ACT,  threadIdx.x); break;
  }
}

// ---------------------------------------------------------------------------
// Kernel 2: fused CommNet forward.
// ---------------------------------------------------------------------------
__global__ void __launch_bounds__(256)
commnet_lstm_kernel(const float* __restrict__ obs,
                    const float* __restrict__ enc_b,
                    const float* __restrict__ msg_b0, const float* __restrict__ msg_b1,
                    const float* __restrict__ b0, const float* __restrict__ b1,
                    const float* __restrict__ act_b,
                    const float* __restrict__ val_W, const float* __restrict__ val_b,
                    const _Float16* __restrict__ wsfr,
                    float* __restrict__ out) {
  // 64KB staging buffer, reused per stage:
  //   enc : [0,16384) obs tile f16 (128x128), [16384,24576) enc frags
  //   msg : [0,4096)  msg frags
  //   gate: [0,16384) Wx frags, [16384,32768) Wh frags
  //   head: [0,1024)  act frags
  __shared__ __align__(16) _Float16 wbuf[32768];
  __shared__ __align__(16) _Float16 hbuf[ROWS_PER_WG][H_D];
  __shared__ __align__(16) _Float16 zbuf[ROWS_PER_WG][H_D];
  __shared__ __align__(16) float    xbuf[ROWS_PER_WG][H_D];  // x = tanh(enc), f32
  __shared__ float Sbuf[4][H_D];

  const int  tid  = threadIdx.x;
  const int  lane = tid & 31;
  const int  wave = tid >> 5;
  const int  m0   = wave * 16;
  const long rowBase = (long)blockIdx.x * ROWS_PER_WG;
  const int  nlo  = lane & 15;           // column within a 16-wide tile
  const int  rhi  = (lane >> 4) << 3;    // +8 rows for upper half-wave

  // ---- stage obs tile (f32 -> f16, coalesced) + fetch enc fragments ---------
  _Float16* obstile = wbuf;
  _Float16* encfr   = wbuf + 16384;
  {
    copy_frags(wsfr + WS_ENC, encfr, 16 * FRAG_ELEMS, tid);
    const float* obsG = obs + rowBase * OBS_D;
    for (int idx = tid * 4; idx < ROWS_PER_WG * OBS_D; idx += NTHREADS * 4) {
      v4f w = *(const v4f*)(obsG + idx);
      v4h hx = {(_Float16)w[0], (_Float16)w[1], (_Float16)w[2], (_Float16)w[3]};
      *(v4h*)(obstile + idx) = hx;
    }
    for (int e = tid; e < ROWS_PER_WG * H_D; e += NTHREADS)
      (&hbuf[0][0])[e] = (_Float16)0.f;
  }
  async_join();
  __syncthreads();

  // ---- x = tanh(obs @ enc_W + enc_b) -> xbuf (f32, LDS) ---------------------
  {
    v16h aobs[4];
#pragma unroll
    for (int kt = 0; kt < 4; ++kt)
      aobs[kt] = load_a_lds(obstile, m0, kt * 32, OBS_D, lane);
#pragma unroll
    for (int nt = 0; nt < 4; ++nt) {
      float bv = enc_b[nt * 16 + nlo];
      v8f acc = {bv, bv, bv, bv, bv, bv, bv, bv};
#pragma unroll
      for (int kt = 0; kt < 4; ++kt)
        acc = wmma_f16(aobs[kt], load_bfrag(encfr, kt * 4 + nt, lane), acc);
#pragma unroll
      for (int e = 0; e < 8; ++e)
        xbuf[m0 + e + rhi][nt * 16 + nlo] = tanh_fast(acc[e]);
    }
  }
  __syncthreads();   // all waves done reading wbuf

  v8f h[4], c[4];
  const v8f vzero = {0.f, 0.f, 0.f, 0.f, 0.f, 0.f, 0.f, 0.f};
#pragma unroll
  for (int nt = 0; nt < 4; ++nt) { h[nt] = vzero; c[nt] = vzero; }

  // ---- 2 communication hops -------------------------------------------------
  for (int hop = 0; hop < 2; ++hop) {
    const float* mB = hop ? msg_b1 : msg_b0;
    const float* lb = hop ? b1 : b0;
    const _Float16* msgF = wsfr + (hop ? WS_MSG1 : WS_MSG0);
    const _Float16* wxF  = wsfr + (hop ? WS_WX1 : WS_WX0);
    const _Float16* whF  = wsfr + (hop ? WS_WH1 : WS_WH0);

    // per-batch column sums; fetch msg fragments
    {
      copy_frags(msgF, wbuf, 8 * FRAG_ELEMS, tid);
      int b = tid >> 6, col = tid & 63;
      float s = 0.f;
#pragma unroll
      for (int j = 0; j < N_AG; ++j) s += (float)hbuf[b * 32 + j][col];
      Sbuf[b][col] = s;
    }
    async_join();
    __syncthreads();

    // comm[r][cc] = (S[batch][cc] - h[r][cc]) / (N-1)  -> zbuf
    {
      const float inv = 1.0f / (float)(N_AG - 1);
      for (int e = tid; e < ROWS_PER_WG * H_D; e += NTHREADS) {
        int r = e >> 6, cc = e & 63;
        zbuf[r][cc] = (_Float16)((Sbuf[r >> 5][cc] - (float)hbuf[r][cc]) * inv);
      }
    }
    __syncthreads();

    // msg = comm @ mW + mb ; z = x + msg
    v8f z[4];
    {
      v16h ca0 = load_a_lds(&zbuf[0][0], m0, 0,  H_D, lane);
      v16h ca1 = load_a_lds(&zbuf[0][0], m0, 32, H_D, lane);
#pragma unroll
      for (int nt = 0; nt < 4; ++nt) {
        float bv = mB[nt * 16 + nlo];
        v8f acc = {bv, bv, bv, bv, bv, bv, bv, bv};
        acc = wmma_f16(ca0, load_bfrag(wbuf, 0 * 4 + nt, lane), acc);
        acc = wmma_f16(ca1, load_bfrag(wbuf, 1 * 4 + nt, lane), acc);
#pragma unroll
        for (int e = 0; e < 8; ++e)
          acc[e] += xbuf[m0 + e + rhi][nt * 16 + nlo];
        z[nt] = acc;
      }
    }
    __syncthreads();   // msg frag reads done before restage

    // stash z (own strip) into zbuf; fetch gate fragments
#pragma unroll
    for (int nt = 0; nt < 4; ++nt)
#pragma unroll
      for (int e = 0; e < 8; ++e)
        zbuf[m0 + e + rhi][nt * 16 + nlo] = (_Float16)z[nt][e];
    copy_frags(wxF, wbuf,         32 * FRAG_ELEMS, tid);
    copy_frags(whF, wbuf + 16384, 32 * FRAG_ELEMS, tid);
    async_join();
    __syncthreads();

    // gates = z @ Wx + h @ Wh + b ; LSTM cell + output nonlinearity
    v16h za0 = load_a_lds(&zbuf[0][0], m0, 0,  H_D, lane);
    v16h za1 = load_a_lds(&zbuf[0][0], m0, 32, H_D, lane);
    v16h ha0 = load_a_lds(&hbuf[0][0], m0, 0,  H_D, lane);
    v16h ha1 = load_a_lds(&hbuf[0][0], m0, 32, H_D, lane);
#pragma unroll
    for (int nt = 0; nt < 4; ++nt) {
      v8f g[4];
#pragma unroll
      for (int q = 0; q < 4; ++q) {
        int ntile = q * 4 + nt;
        float bv = lb[ntile * 16 + nlo];
        v8f acc = {bv, bv, bv, bv, bv, bv, bv, bv};
        acc = wmma_f16(za0, load_bfrag(wbuf,         0 * 16 + ntile, lane), acc);
        acc = wmma_f16(za1, load_bfrag(wbuf,         1 * 16 + ntile, lane), acc);
        acc = wmma_f16(ha0, load_bfrag(wbuf + 16384, 0 * 16 + ntile, lane), acc);
        acc = wmma_f16(ha1, load_bfrag(wbuf + 16384, 1 * 16 + ntile, lane), acc);
        g[q] = acc;
      }
#pragma unroll
      for (int e = 0; e < 8; ++e) {
        float gi = sigf(g[0][e]);
        float gf = sigf(g[1][e]);
        float gg = tanh_fast(g[2][e]);
        float go = sigf(g[3][e]);
        float cn = gf * c[nt][e] + gi * gg;
        float hn = go * tanh_fast(cn);
        c[nt][e] = cn;
        h[nt][e] = tanh_fast(hn + z[nt][e]);
      }
    }
    __syncthreads();

    // publish new h (f16)
#pragma unroll
    for (int nt = 0; nt < 4; ++nt)
#pragma unroll
      for (int e = 0; e < 8; ++e)
        hbuf[m0 + e + rhi][nt * 16 + nlo] = (_Float16)h[nt][e];
    __syncthreads();
  }

  // ---- logits head ----------------------------------------------------------
  copy_frags(wsfr + WS_ACT, wbuf, 2 * FRAG_ELEMS, tid);
  async_join();
  __syncthreads();
  {
    v16h ha0 = load_a_lds(&hbuf[0][0], m0, 0,  H_D, lane);
    v16h ha1 = load_a_lds(&hbuf[0][0], m0, 32, H_D, lane);
    float bv = act_b[nlo];
    v8f acc = {bv, bv, bv, bv, bv, bv, bv, bv};
    acc = wmma_f16(ha0, load_bfrag(wbuf, 0, lane), acc);
    acc = wmma_f16(ha1, load_bfrag(wbuf, 1, lane), acc);
#pragma unroll
    for (int e = 0; e < 8; ++e) {
      long r = rowBase + m0 + e + rhi;
      out[OFF_L + r * A_D + nlo] = acc[e];
    }
  }

  // ---- values head: dot(h_row, val_W) via register tiles + lane reduce ------
  {
    float vw[4];
#pragma unroll
    for (int nt = 0; nt < 4; ++nt) vw[nt] = val_W[nt * 16 + nlo];
    float vb = val_b[0];
#pragma unroll
    for (int e = 0; e < 8; ++e) {
      float p = 0.f;
#pragma unroll
      for (int nt = 0; nt < 4; ++nt) p += h[nt][e] * vw[nt];
#pragma unroll
      for (int msk = 1; msk < 16; msk <<= 1) p += __shfl_xor(p, msk, 32);
      if (nlo == 0) out[OFF_V + rowBase + m0 + e + rhi] = p + vb;
    }
  }

  // ---- h, c outputs in f32 from registers -----------------------------------
#pragma unroll
  for (int nt = 0; nt < 4; ++nt)
#pragma unroll
    for (int e = 0; e < 8; ++e) {
      long r = rowBase + m0 + e + rhi;
      int  cc = nt * 16 + nlo;
      out[OFF_H + r * H_D + cc] = h[nt][e];
      out[OFF_C + r * H_D + cc] = c[nt][e];
    }
}

extern "C" void kernel_launch(void* const* d_in, const int* in_sizes, int n_in,
                              void* d_out, int out_size, void* d_ws, size_t ws_size,
                              hipStream_t stream) {
  (void)in_sizes; (void)n_in; (void)out_size; (void)ws_size;
  const float* obs    = (const float*)d_in[0];
  const float* enc_W  = (const float*)d_in[1];
  const float* enc_b  = (const float*)d_in[2];
  const float* msg_W0 = (const float*)d_in[3];
  const float* msg_b0 = (const float*)d_in[4];
  const float* msg_W1 = (const float*)d_in[5];
  const float* msg_b1 = (const float*)d_in[6];
  const float* Wx0    = (const float*)d_in[7];
  const float* Wh0    = (const float*)d_in[8];
  const float* b0     = (const float*)d_in[9];
  const float* Wx1    = (const float*)d_in[10];
  const float* Wh1    = (const float*)d_in[11];
  const float* b1     = (const float*)d_in[12];
  const float* act_W  = (const float*)d_in[13];
  const float* act_b  = (const float*)d_in[14];
  const float* val_W  = (const float*)d_in[15];
  const float* val_b  = (const float*)d_in[16];
  float* out = (float*)d_out;
  _Float16* ws = (_Float16*)d_ws;

  hipLaunchKernelGGL(stage_weights_kernel, dim3(8), dim3(NTHREADS), 0, stream,
                     enc_W, msg_W0, msg_W1, Wx0, Wh0, Wx1, Wh1, act_W, ws);
  hipLaunchKernelGGL(commnet_lstm_kernel, dim3(BN / ROWS_PER_WG), dim3(NTHREADS),
                     0, stream,
                     obs, enc_b, msg_b0, msg_b1, b0, b1, act_b,
                     val_W, val_b, (const _Float16*)ws, out);
}
